// KernelBAE_43911745635093
// MI455X (gfx1250) — 8-bit matrix operand order repaired and checked by reading, compiled, not run on hardware
//
#include <hip/hip_runtime.h>


#define N_ROWS 4096
#define M_FEAT 128
#define NWORDS 128
#define BETA   0.01f
#define TEMP_  0.5f
#define NF     4096.0f
#define TCONST (4095.0f / 4096.0f)

#define GIBBS_LDS_BYTES 222272

__global__ __launch_bounds__(1024, 1)
void bae_gibbs_sweep(const float* __restrict__ Kin,
                     const float* __restrict__ Sin,
                     const float* __restrict__ Uin,
                     const int*   __restrict__ perm,
                     unsigned char* __restrict__ S8out) {
  extern __shared__ unsigned char smem_raw[];
  float*        StS      = (float*)smem_raw;
  float*        Rm       = StS + 16384;
  unsigned int* Sbits    = (unsigned int*)(Rm + 16384);
  float*        kld      = (float*)(Sbits + 16384);
  float*        partials = kld + 4096;
  float*        St1      = partials + 1024;
  float*        Sk       = St1 + 128;
  float*        sbarA    = Sk + 128;
  float*        hA       = sbarA + 128;
  float*        JiiA     = hA + 128;
  float*        newsA    = JiiA + 128;
  float*        srow     = newsA + 128;
  float*        rarr     = srow + 128;
  float*        tmpA     = rarr + 128;
  float*        tmpB     = tmpA + 128;
  float*        sc       = tmpB + 128;

  const int tid  = threadIdx.x;
  const int j    = tid & 127;
  const int g    = tid >> 7;

  #pragma unroll 1
  for (int idx = tid; idx < 16384; idx += 1024) {
    const int col = idx >> 7, w = idx & 127;
    unsigned int bits = 0u;
    #pragma unroll 1
    for (int b = 0; b < 32; ++b) {
      const int n = w * 32 + b;
      bits |= (Sin[(size_t)n * M_FEAT + col] != 0.0f ? 1u : 0u) << b;
    }
    Sbits[idx] = bits;
  }
  __syncthreads();

  #pragma unroll 1
  for (int e = 0; e < 16; ++e) {
    const int a = g * 16 + e;
    int cnt = 0;
    #pragma unroll 1
    for (int w = 0; w < NWORDS; ++w)
      cnt += __popc(Sbits[a * NWORDS + w] & Sbits[j * NWORDS + w]);
    StS[a * M_FEAT + j] = (float)cnt;
  }
  if (tid < 128) {
    int cnt = 0;
    #pragma unroll 1
    for (int w = 0; w < NWORDS; ++w) cnt += __popc(Sbits[tid * NWORDS + w]);
    St1[tid] = (float)cnt;
  }
  __syncthreads();

  #pragma unroll 1
  for (int t = 0; t < N_ROWS; ++t) {
    const int i = perm[t];
    const float4* Kr4 = (const float4*)(Kin + (size_t)i * N_ROWS);
    ((float4*)kld)[tid] = Kr4[tid];
    if (t + 1 < N_ROWS)
      __builtin_prefetch(Kin + (size_t)perm[t + 1] * N_ROWS, 0, 0);
    __syncthreads();
    const float k0 = kld[i];

    if (tid < 128)
      srow[tid] = (float)((Sbits[tid * NWORDS + (i >> 5)] >> (i & 31)) & 1u);

    {
      float acc = 0.0f;
      const int wbase = g * 16;
      #pragma unroll 1
      for (int w = 0; w < 16; ++w) {
        const unsigned int mbits = Sbits[j * NWORDS + wbase + w];
        const float* kk = kld + (wbase + w) * 32;
        #pragma unroll 1
        for (int b = 0; b < 32; ++b)
          acc += ((mbits >> b) & 1u) ? kk[b] : 0.0f;
      }
      partials[tid] = acc;
    }
    __syncthreads();

    if (tid < 128) {
      float sk = 0.0f;
      #pragma unroll
      for (int gg = 0; gg < 8; ++gg) sk += partials[gg * 128 + j];
      Sk[j]  = sk - srow[j] * k0;
      St1[j] = St1[j] - srow[j];
    }
    __syncthreads();

    {
      float rpart = 0.0f;
      const float sb = srow[j];
      const float St1b = St1[j];
      #pragma unroll 1
      for (int e = 0; e < 16; ++e) {
        const int a   = g * 16 + e;
        const int idx = a * M_FEAT + j;
        const float sts = StS[idx] - srow[a] * sb;
        StS[idx] = sts;
        const float St1a = St1[a];
        const float d1 = sts;
        const float d2 = St1b - sts;
        const float d3 = St1a - sts;
        const float d4 = (NF - 1.0f) - St1a - St1b + sts;
        const float b1 = (d1 < d2 && d1 < d3 && d1 < d4) ? 1.0f : 0.0f;
        const float b2 = (d2 < d1 && d2 < d3 && d2 < d4) ? 1.0f : 0.0f;
        const float b3 = (d3 < d2 && d3 < d1 && d3 < d4) ? 1.0f : 0.0f;
        const float b4 = (d4 < d2 && d4 < d3 && d4 < d1) ? 1.0f : 0.0f;
        Rm[idx] = b1 - b2 - b3 + b4;
        rpart  += b2 - b4;
      }
      partials[tid] = rpart;
    }
    __syncthreads();

    if (tid < 128) {
      float rs = 0.0f;
      #pragma unroll
      for (int gg = 0; gg < 8; ++gg) rs += partials[gg * 128 + j];
      rarr[j] = rs;
      const float sbar = St1[j] * (1.0f / (NF - 1.0f));
      sbarA[j] = sbar;
      const float uv  = 2.0f * sbar - 1.0f;
      const float ssc = sbar * (1.0f - sbar);
      tmpA[j] = ssc;
      tmpB[j] = sbar * (srow[j] - sbar);
      JiiA[j] = 2.0f * (NF - 1.0f) * ssc + TCONST * uv * uv;
    }
    __syncthreads();

    if (tid == 0) {
      float ssc_sum = 0.f, sx0 = 0.f, sum_s = 0.f, sum_sb = 0.f;
      #pragma unroll 1
      for (int q = 0; q < 128; ++q) {
        ssc_sum += tmpA[q]; sx0 += tmpB[q];
        sum_s += srow[q];   sum_sb += sbarA[q];
      }
      sc[0] = sx0;
      sc[1] = 2.0f * sx0 - sum_s + sum_sb;
      sc[2] = ssc_sum;
    }
    __syncthreads();

    if (tid < 128) {
      const float uv = 2.0f * sbarA[j] - 1.0f;
      hA[j] = TCONST * (sc[2] - k0) * uv + 2.0f * Sk[j] - BETA * rarr[j];
    }
    __syncthreads();

    if (tid < 32) {
      const int lane = tid;
      float nl0 = srow[lane*4+0], nl1 = srow[lane*4+1];
      float nl2 = srow[lane*4+2], nl3 = srow[lane*4+3];
      const float sb0 = sbarA[lane*4+0], sb1 = sbarA[lane*4+1];
      const float sb2 = sbarA[lane*4+2], sb3 = sbarA[lane*4+3];
      float sx = sc[0], ux = sc[1];
      const float* uRow = Uin + (size_t)t * M_FEAT;
      #pragma unroll 1
      for (int jj = 0; jj < 128; ++jj) {
        const int owner = jj >> 2, c = jj & 3;
        const float nlc   = (c==0)?nl0:(c==1)?nl1:(c==2)?nl2:nl3;
        const float oldnj = __shfl(nlc, owner);
        const float4 sr = ((const float4*)(StS + jj * M_FEAT))[lane];
        const float4 rr = ((const float4*)(Rm  + jj * M_FEAT))[lane];
        float d1 = sr.x*(nl0-sb0) + sr.y*(nl1-sb1) + sr.z*(nl2-sb2) + sr.w*(nl3-sb3);
        float d2 = rr.x*nl0 + rr.y*nl1 + rr.z*nl2 + rr.w*nl3;
        #pragma unroll
        for (int mk = 16; mk; mk >>= 1) {
          d1 += __shfl_xor(d1, mk);
          d2 += __shfl_xor(d2, mk);
        }
        const float s_j = sbarA[jj];
        const float uvj = 2.0f * s_j - 1.0f;
        const float Jj  = JiiA[jj];
        const float dot = 2.0f * d1 - 2.0f * (NF - 1.0f) * s_j * sx
                        + TCONST * uvj * ux - Jj * oldnj + BETA * d2;
        const float curr = (hA[jj] - 0.5f * Jj - dot) * (1.0f / TEMP_);
        const float prob = (curr < -100.0f) ? 0.0f
                         : (curr >  100.0f) ? 1.0f
                         : 1.0f / (1.0f + __expf(-curr));
        const float sj = (uRow[jj] < prob) ? 1.0f : 0.0f;
        const float ds = sj - oldnj;
        if (lane == owner) {
          if      (c == 0) nl0 = sj;
          else if (c == 1) nl1 = sj;
          else if (c == 2) nl2 = sj;
          else             nl3 = sj;
        }
        sx += ds * s_j;
        ux += ds * uvj;
      }
      newsA[lane*4+0] = nl0; newsA[lane*4+1] = nl1;
      newsA[lane*4+2] = nl2; newsA[lane*4+3] = nl3;
    }
    __syncthreads();

    if (tid < 128) {
      const float nj = newsA[tid];
      St1[tid] += nj;
      unsigned int w = Sbits[tid * NWORDS + (i >> 5)];
      const unsigned int bit = 1u << (i & 31);
      w = (nj != 0.0f) ? (w | bit) : (w & ~bit);
      Sbits[tid * NWORDS + (i >> 5)] = w;
    }
    {
      const float nb = newsA[j];
      #pragma unroll 1
      for (int e = 0; e < 16; ++e) {
        const int a = g * 16 + e;
        StS[a * M_FEAT + j] += newsA[a] * nb;
      }
    }
    __syncthreads();
  }

  #pragma unroll 1
  for (int idx = tid; idx < N_ROWS * M_FEAT; idx += 1024) {
    const int n = idx >> 7, col = idx & 127;
    S8out[idx] = (unsigned char)((Sbits[col * NWORDS + (n >> 5)] >> (n & 31)) & 1u);
  }
}

typedef __attribute__((ext_vector_type(8))) int v8i;

__global__ __launch_bounds__(256)
void bae_ssT_wmma(const unsigned char* __restrict__ S8,
                  float* __restrict__ out) {
  const int lane = threadIdx.x & 31;
  const int wv   = threadIdx.x >> 5;
  const int tile = blockIdx.x * 8 + wv;
  const int ib   = (tile >> 8) * 16;
  const int jb   = (tile & 255) * 16;
  const int m    = lane & 15;
  const bool hi  = lane >= 16;

  const unsigned int* ra = (const unsigned int*)(S8 + (size_t)(ib + m) * M_FEAT);
  const unsigned int* rb = (const unsigned int*)(S8 + (size_t)(jb + m) * M_FEAT);

  v8i A0, A1, B0, B1;
  const int aw0 = hi ? 2 : 0;
  #pragma unroll
  for (int v = 0; v < 8; ++v) {
    const int w = aw0 + (v & 1) + ((v >> 1) << 2);
    A0[v] = (int)ra[w];
    A1[v] = (int)ra[w + 16];
  }
  const int bw0 = hi ? 2 : 0;
  #pragma unroll
  for (int v = 0; v < 8; ++v) {
    const int w = bw0 + (v & 1) + ((v >> 1) << 2);
    B0[v] = (int)rb[w];
    B1[v] = (int)rb[w + 16];
  }

  v8i acc = {};
  acc = __builtin_amdgcn_wmma_i32_16x16x64_iu8(false, A0, false, B0, acc, false, false);
  acc = __builtin_amdgcn_wmma_i32_16x16x64_iu8(false, A1, false, B1, acc, false, false);

  const int col = jb + m;
  #pragma unroll
  for (int v = 0; v < 8; ++v) {
    const int row = ib + v + (hi ? 8 : 0);
    out[(size_t)row * N_ROWS + col] = (float)acc[v];
  }
}

extern "C" void kernel_launch(void* const* d_in, const int* in_sizes, int n_in,
                              void* d_out, int out_size, void* d_ws, size_t ws_size,
                              hipStream_t stream) {
  (void)in_sizes; (void)n_in; (void)out_size; (void)ws_size;
  const float* Kmat = (const float*)d_in[0];
  const float* S    = (const float*)d_in[1];
  const float* U    = (const float*)d_in[2];
  const int*   perm = (const int*)d_in[3];
  float* out = (float*)d_out;
  unsigned char* S8 = (unsigned char*)d_ws;

  bae_gibbs_sweep<<<1, 1024, GIBBS_LDS_BYTES, stream>>>(Kmat, S, U, perm, S8);
  bae_ssT_wmma<<<(256 * 256) / 8, 256, 0, stream>>>(S8, out);
}
